// RecurrentGCN_14474039788131
// MI455X (gfx1250) — matrix operand order repaired and checked by reading, not compiled, not run on hardware
//
#include <hip/hip_runtime.h>

typedef __attribute__((ext_vector_type(16))) __bf16 v16bf;
typedef __attribute__((ext_vector_type(8)))  float  v8f;
typedef __attribute__((ext_vector_type(4)))  float  f32x4;

#define NNODES 50000
#define HID    256
#define NH     ((size_t)NNODES * HID)   // 12,800,000 elements

__device__ __forceinline__ float fsig(float x)  { return 1.0f / (1.0f + __expf(-x)); }
__device__ __forceinline__ float ftanh(float x) { return 2.0f / (1.0f + __expf(-2.0f * x)) - 1.0f; }

__device__ __forceinline__ v8f wmma_bf16(v16bf a, v16bf b, v8f c) {
    return __builtin_amdgcn_wmma_f32_16x16x32_bf16(false, a, false, b, (short)0, c, false, false);
}

// A-fragment (16x32 bf16) per-lane load from f32 row-major source.
// lane<16: K 0-15 ; lane>=16: K 16-31 (caller adds c0 = half*16)
__device__ __forceinline__ v16bf load_a(const float* __restrict__ p) {
    f32x4 a0 = *(const f32x4*)(p);
    f32x4 a1 = *(const f32x4*)(p + 4);
    f32x4 a2 = *(const f32x4*)(p + 8);
    f32x4 a3 = *(const f32x4*)(p + 12);
    v16bf af;
    af[0]  = (__bf16)a0[0]; af[1]  = (__bf16)a0[1]; af[2]  = (__bf16)a0[2]; af[3]  = (__bf16)a0[3];
    af[4]  = (__bf16)a1[0]; af[5]  = (__bf16)a1[1]; af[6]  = (__bf16)a1[2]; af[7]  = (__bf16)a1[3];
    af[8]  = (__bf16)a2[0]; af[9]  = (__bf16)a2[1]; af[10] = (__bf16)a2[2]; af[11] = (__bf16)a2[3];
    af[12] = (__bf16)a3[0]; af[13] = (__bf16)a3[1]; af[14] = (__bf16)a3[2]; af[15] = (__bf16)a3[3];
    return af;
}

__device__ __forceinline__ v16bf load_a_relu(const float* __restrict__ p) {
    f32x4 a0 = *(const f32x4*)(p);
    f32x4 a1 = *(const f32x4*)(p + 4);
    f32x4 a2 = *(const f32x4*)(p + 8);
    f32x4 a3 = *(const f32x4*)(p + 12);
    v16bf af;
    af[0]  = (__bf16)fmaxf(a0[0], 0.f); af[1]  = (__bf16)fmaxf(a0[1], 0.f);
    af[2]  = (__bf16)fmaxf(a0[2], 0.f); af[3]  = (__bf16)fmaxf(a0[3], 0.f);
    af[4]  = (__bf16)fmaxf(a1[0], 0.f); af[5]  = (__bf16)fmaxf(a1[1], 0.f);
    af[6]  = (__bf16)fmaxf(a1[2], 0.f); af[7]  = (__bf16)fmaxf(a1[3], 0.f);
    af[8]  = (__bf16)fmaxf(a2[0], 0.f); af[9]  = (__bf16)fmaxf(a2[1], 0.f);
    af[10] = (__bf16)fmaxf(a2[2], 0.f); af[11] = (__bf16)fmaxf(a2[3], 0.f);
    af[12] = (__bf16)fmaxf(a3[0], 0.f); af[13] = (__bf16)fmaxf(a3[1], 0.f);
    af[14] = (__bf16)fmaxf(a3[2], 0.f); af[15] = (__bf16)fmaxf(a3[3], 0.f);
    return af;
}

// ---------------- prep: pack weights to bf16, K-contiguous per output column ----
__global__ void prep_wt(const float* __restrict__ Wi, const float* __restrict__ Wf,
                        const float* __restrict__ Wc, const float* __restrict__ Wo,
                        const float* __restrict__ Ti, const float* __restrict__ Tf,
                        const float* __restrict__ Tc, const float* __restrict__ To,
                        __bf16* __restrict__ Wt) {
    int idx = blockIdx.x * blockDim.x + threadIdx.x;   // 1024*512 threads
    int n = idx >> 9, k = idx & 511;
    int g = n >> 8, j = n & 255;
    const float* Wg[4] = {Wi, Wf, Wc, Wo};
    const float* Tg[4] = {Ti, Tf, Tc, To};
    float v = (k < 256) ? Wg[g][(size_t)k * 256 + j] : Tg[g][(size_t)(k - 256) * 256 + j];
    Wt[idx] = (__bf16)v;
}

__global__ void prep_wlin(const float* __restrict__ Wlin, __bf16* __restrict__ WlinT) {
    int idx = blockIdx.x * blockDim.x + threadIdx.x;   // 256*256 threads
    int n = idx >> 8, k = idx & 255;
    WlinT[idx] = (__bf16)Wlin[(size_t)k * 256 + n];
}

__global__ void prep_bias(const float* __restrict__ bthi, const float* __restrict__ bthf,
                          const float* __restrict__ bthc, const float* __restrict__ btho,
                          const float* __restrict__ bi,   const float* __restrict__ bf,
                          const float* __restrict__ bc,   const float* __restrict__ bo,
                          float* __restrict__ bias4) {
    int idx = blockIdx.x * blockDim.x + threadIdx.x;   // 1024 threads
    int g = idx >> 8, j = idx & 255;
    const float* bth[4] = {bthi, bthf, bthc, btho};
    const float* bb[4]  = {bi, bf, bc, bo};
    bias4[idx] = bth[g][j] + bb[g][j];
}

// ---- in-register LSTM gate epilogue for one 16-row tile -----------------------
__device__ __forceinline__ void gate_epi(
    const v8f* __restrict__ acc, int rowbase, int w, int half, int m,
    const float* __restrict__ c, const float* __restrict__ bias4,
    const float* __restrict__ wci, const float* __restrict__ wcf, const float* __restrict__ wco,
    float* __restrict__ h0_out, float* __restrict__ c0_out) {
    #pragma unroll
    for (int s = 0; s < 2; ++s) {
        #pragma unroll
        for (int v = 0; v < 8; ++v) {
            const int row = rowbase + v + 8 * half;
            const int col = w * 32 + s * 16 + m;
            const float cv = c[(size_t)row * 256 + col];
            float I = fsig (acc[0 + s][v] + bias4[col]       + wci[col] * cv);
            float F = fsig (acc[2 + s][v] + bias4[256 + col] + wcf[col] * cv);
            float T = ftanh(acc[4 + s][v] + bias4[512 + col]);
            float Cn = F * cv + I * T;
            float O = fsig (acc[6 + s][v] + bias4[768 + col] + wco[col] * Cn);
            float Hn = O * ftanh(Cn);
            h0_out[(size_t)row * 256 + col] = Hn;
            c0_out[(size_t)row * 256 + col] = Cn;
        }
    }
}

// ---------------- fused GC-LSTM cell: 32-row blocks, B reused for 2 M-tiles ----
__global__ __launch_bounds__(256) void gclstm_gemm(
    const float* __restrict__ x, const float* __restrict__ h, const float* __restrict__ c,
    const __bf16* __restrict__ Wt, const float* __restrict__ bias4,
    const float* __restrict__ wci, const float* __restrict__ wcf, const float* __restrict__ wco,
    float* __restrict__ h0_out, float* __restrict__ c0_out) {

    const int w    = threadIdx.x >> 5;     // wave 0..7 : owns cols 32w..32w+31 of all 4 gates
    const int lane = threadIdx.x & 31;
    const int half = lane >> 4;
    const int m    = lane & 15;
    const int row0 = blockIdx.x * 32;
    const bool has2 = (row0 + 16) < NNODES;        // uniform; last block has 16 rows only
    const int c0   = half * 16;
    const int rA0  = row0 + m;
    const int rA1  = min(row0 + 16 + m, NNODES - 1);  // clamp: keep loads in bounds, EXEC all-ones

    v8f acc0[8] = {};
    v8f acc1[8] = {};

    for (int ks = 0; ks < 16; ++ks) {
        const float* base = (ks < 8) ? x : h;
        const int kb = (ks & 7) * 32 + c0;
        v16bf af0 = load_a(base + (size_t)rA0 * 256 + kb);
        v16bf af1 = load_a(base + (size_t)rA1 * 256 + kb);

        const int kb2 = ks * 32 + half * 16;
        #pragma unroll
        for (int g = 0; g < 4; ++g) {
            #pragma unroll
            for (int s = 0; s < 2; ++s) {
                const int n0 = g * 256 + w * 32 + s * 16;
                v16bf bfrag = *(const v16bf*)(Wt + (size_t)(n0 + m) * 512 + kb2);
                acc0[g * 2 + s] = wmma_bf16(af0, bfrag, acc0[g * 2 + s]);
                acc1[g * 2 + s] = wmma_bf16(af1, bfrag, acc1[g * 2 + s]);
            }
        }
    }

    gate_epi(acc0, row0, w, half, m, c, bias4, wci, wcf, wco, h0_out, c0_out);
    if (has2)
        gate_epi(acc1, row0 + 16, w, half, m, c, bias4, wci, wcf, wco, h0_out, c0_out);
}

// ---------------- out = relu(h0) @ W_lin + b_lin : 32-row blocks ---------------
__global__ __launch_bounds__(256) void lin_gemm(
    const float* __restrict__ h0, const __bf16* __restrict__ WlinT,
    const float* __restrict__ blin, float* __restrict__ out) {

    const int w    = threadIdx.x >> 5;
    const int lane = threadIdx.x & 31;
    const int half = lane >> 4;
    const int m    = lane & 15;
    const int row0 = blockIdx.x * 32;
    const bool has2 = (row0 + 16) < NNODES;
    const int c0   = half * 16;
    const int rA0  = row0 + m;
    const int rA1  = min(row0 + 16 + m, NNODES - 1);

    v8f acc0[2] = {};
    v8f acc1[2] = {};

    #pragma unroll
    for (int ks = 0; ks < 8; ++ks) {
        const int kb = ks * 32 + c0;
        v16bf af0 = load_a_relu(h0 + (size_t)rA0 * 256 + kb);
        v16bf af1 = load_a_relu(h0 + (size_t)rA1 * 256 + kb);

        const int kb2 = ks * 32 + half * 16;
        #pragma unroll
        for (int t = 0; t < 2; ++t) {
            const int n0 = (2 * w + t) * 16;
            v16bf bfrag = *(const v16bf*)(WlinT + (size_t)(n0 + m) * 256 + kb2);
            acc0[t] = wmma_bf16(af0, bfrag, acc0[t]);
            acc1[t] = wmma_bf16(af1, bfrag, acc1[t]);
        }
    }

    #pragma unroll
    for (int t = 0; t < 2; ++t) {
        const int col = (2 * w + t) * 16 + m;
        const float bl = blin[col];
        #pragma unroll
        for (int v = 0; v < 8; ++v) {
            const int row = row0 + v + 8 * half;
            out[(size_t)row * 256 + col] = acc0[t][v] + bl;
        }
        if (has2) {
            #pragma unroll
            for (int v = 0; v < 8; ++v) {
                const int row = row0 + 16 + v + 8 * half;
                out[(size_t)row * 256 + col] = acc1[t][v] + bl;
            }
        }
    }
}

extern "C" void kernel_launch(void* const* d_in, const int* in_sizes, int n_in,
                              void* d_out, int out_size, void* d_ws, size_t ws_size,
                              hipStream_t stream) {
    const float* x    = (const float*)d_in[0];
    const float* h    = (const float*)d_in[3];
    const float* c    = (const float*)d_in[4];
    const float* Wi   = (const float*)d_in[5];
    const float* Wf   = (const float*)d_in[6];
    const float* Wc   = (const float*)d_in[7];
    const float* Wo   = (const float*)d_in[8];
    const float* Ti   = (const float*)d_in[9];
    const float* Tf   = (const float*)d_in[10];
    const float* Tc   = (const float*)d_in[11];
    const float* To   = (const float*)d_in[12];
    const float* bthi = (const float*)d_in[13];
    const float* bthf = (const float*)d_in[14];
    const float* bthc = (const float*)d_in[15];
    const float* btho = (const float*)d_in[16];
    const float* wci  = (const float*)d_in[17];
    const float* wcf  = (const float*)d_in[18];
    const float* wco  = (const float*)d_in[19];
    const float* bi   = (const float*)d_in[20];
    const float* bf   = (const float*)d_in[21];
    const float* bc   = (const float*)d_in[22];
    const float* bo   = (const float*)d_in[23];
    const float* Wlin = (const float*)d_in[24];
    const float* blin = (const float*)d_in[25];

    // workspace layout: Wt (1MB bf16) | WlinT (128KB bf16) | bias4 (4KB f32)
    __bf16* Wt    = (__bf16*)d_ws;
    __bf16* WlinT = Wt + 1024 * 512;
    float*  bias4 = (float*)(WlinT + 256 * 256);

    float* out = (float*)d_out;
    float* h0o = out + NH;
    float* c0o = out + 2 * NH;

    prep_wt  <<<(1024 * 512) / 256, 256, 0, stream>>>(Wi, Wf, Wc, Wo, Ti, Tf, Tc, To, Wt);
    prep_wlin<<<(256 * 256) / 256, 256, 0, stream>>>(Wlin, WlinT);
    prep_bias<<<4, 256, 0, stream>>>(bthi, bthf, bthc, btho, bi, bf, bc, bo, bias4);

    const int nblk = (NNODES + 31) / 32;   // 1563
    gclstm_gemm<<<nblk, 256, 0, stream>>>(x, h, c, Wt, bias4, wci, wcf, wco, h0o, c0o);
    lin_gemm   <<<nblk, 256, 0, stream>>>(h0o, WlinT, blin, out);
}
